// Llama_64287070486900
// MI455X (gfx1250) — matrix operand order repaired and checked by reading, not compiled, not run on hardware
//
#include <hip/hip_runtime.h>
#include <stdint.h>

// ---------------- model constants ----------------
#define LAYERS 4
#define DMODEL 2048
#define HQ     16
#define HKV    4
#define HD     128
#define PFFN   8192
#define VOCAB  32000
#define BATCH  2
#define SEQ    1024
#define MTOT   (BATCH * SEQ)           // 2048 token rows
#define QKVW   (DMODEL + 2 * HKV * HD) // 3072
#define RATIO  (HQ / HKV)              // 4
#define THETA  500000.0f
#define EPSV   1e-5f

// ---------------- WMMA vector types ----------------
typedef __attribute__((ext_vector_type(16))) __bf16        v16bf;
typedef __attribute__((ext_vector_type(8)))  float         v8f;
typedef __attribute__((ext_vector_type(8)))  unsigned int  v8u;

// gfx1250 async global->LDS path (ASYNCcnt-tracked), guarded for portability
#if __has_builtin(__builtin_amdgcn_global_load_async_to_lds_b128) && \
    __has_builtin(__builtin_amdgcn_s_wait_asynccnt)
#define HAVE_ASYNC_LDS 1
typedef int v4i __attribute__((vector_size(16)));
typedef __attribute__((address_space(1))) v4i as1_v4i;   // global 128-bit chunk
typedef __attribute__((address_space(3))) v4i as3_v4i;   // LDS 128-bit chunk
#else
#define HAVE_ASYNC_LDS 0
#endif

static __device__ __forceinline__ v16bf make_frag(uint4 a, uint4 b) {
    v8u u;
    u[0] = a.x; u[1] = a.y; u[2] = a.z; u[3] = a.w;
    u[4] = b.x; u[5] = b.y; u[6] = b.z; u[7] = b.w;
    return __builtin_bit_cast(v16bf, u);
}

// ---------------- embedding gather ----------------
__global__ __launch_bounds__(256) void embed_kernel(
    const int* __restrict__ tokens, const float* __restrict__ emb,
    float* __restrict__ x) {
    const int row = blockIdx.x;
    const int t   = tokens[row];
    const float* src = emb + (size_t)t * DMODEL;
    float* dst = x + (size_t)row * DMODEL;
    for (int c = threadIdx.x; c < DMODEL; c += 256) dst[c] = src[c];
}

// ---------------- rmsnorm: fp32 in -> bf16 out ----------------
__global__ __launch_bounds__(256) void rmsnorm_kernel(
    const float* __restrict__ x, const float* __restrict__ w,
    __bf16* __restrict__ out) {
    __shared__ float red[256];
    const int row = blockIdx.x;
    const int tid = threadIdx.x;
    const float* xr = x + (size_t)row * DMODEL;
    float ss = 0.f;
    for (int c = tid; c < DMODEL; c += 256) { float v = xr[c]; ss += v * v; }
    red[tid] = ss;
    __syncthreads();
    for (int st = 128; st > 0; st >>= 1) {
        if (tid < st) red[tid] += red[tid + st];
        __syncthreads();
    }
    const float rinv = rsqrtf(red[0] / (float)DMODEL + EPSV);
    __bf16* o = out + (size_t)row * DMODEL;
    for (int c = tid; c < DMODEL; c += 256) o[c] = (__bf16)(w[c] * xr[c] * rinv);
}

// ---------------- WMMA GEMM ----------------
// C[M,N] = A[M,K](bf16) * W[K,N](fp32 -> bf16 in LDS), fp32 accumulate.
// mode 0: Cf = acc
// mode 1: Cf = aux + acc            (residual add; aux may alias Cf)
// mode 2: Cb = bf16(silu(aux)*acc)  (fused SwiGLU)
#define BM 128
#define BN 64
#define BK 32
#define SAS 40   // LDS stride (bf16 elems) for A tile rows (padded)
#define SWS 40   // LDS stride (bf16 elems) for W^T tile cols (padded)

__global__ __launch_bounds__(256) void gemm_bf16_wmma(
    const __bf16* __restrict__ A, int lda,
    const float* __restrict__ W, int ldw,
    float* __restrict__ Cf, __bf16* __restrict__ Cb, int ldc,
    int K, int mode, const float* __restrict__ aux, int ldaux) {
    __shared__ __bf16 sA[BM * SAS];
    __shared__ __bf16 sW[BN * SWS];

    const int tid  = threadIdx.x;
    const int wave = tid >> 5;
    const int lane = tid & 31;
    const int lr   = lane & 15;
    const int hi   = lane >> 4;           // 0 or 1
    const int m0   = blockIdx.y * BM;
    const int n0   = blockIdx.x * BN;

    const int rowA = (wave << 4) + lr;    // wave-local A row in tile
    const int kbA  = hi ? 8 : 0;          // A frag K base (ISA 16-bit A layout)
    const int kbB  = hi ? 8 : 0;          // B frag K base (ISA 16-bit B layout)

    v8f acc[4] = {};

    for (int k0 = 0; k0 < K; k0 += BK) {
        // ---- stage A tile: BM x BK bf16 ----
#if HAVE_ASYNC_LDS
        // async global->LDS copy, 16B per lane per op, tracked on ASYNCcnt
#pragma unroll
        for (int i = 0; i < 2; ++i) {
            const int idx = tid + i * 256;
            const int r   = idx >> 2;
            const int c   = (idx & 3) << 3;
            const __bf16* gsrc = A + (size_t)(m0 + r) * lda + k0 + c;
            __builtin_amdgcn_global_load_async_to_lds_b128(
                (as1_v4i*)(uintptr_t)gsrc,
                (as3_v4i*)(uintptr_t)(unsigned int)(uintptr_t)&sA[r * SAS + c],
                0, 0);
        }
#else
#pragma unroll
        for (int i = 0; i < 2; ++i) {
            const int idx = tid + i * 256;
            const int r   = idx >> 2;
            const int c   = (idx & 3) << 3;
            const uint4 v = *reinterpret_cast<const uint4*>(
                A + (size_t)(m0 + r) * lda + k0 + c);
            *reinterpret_cast<uint4*>(&sA[r * SAS + c]) = v;
        }
#endif
        // ---- stage W tile transposed & converted: sW[col][k] bf16 ----
        {
            const int r  = tid >> 3;          // 0..31 : K row
            const int cg = (tid & 7) << 3;    // 0..56 : col group of 8
            const float* wp = W + (size_t)(k0 + r) * ldw + n0 + cg;
            const float4 f0 = *reinterpret_cast<const float4*>(wp);
            const float4 f1 = *reinterpret_cast<const float4*>(wp + 4);
            sW[(cg + 0) * SWS + r] = (__bf16)f0.x;
            sW[(cg + 1) * SWS + r] = (__bf16)f0.y;
            sW[(cg + 2) * SWS + r] = (__bf16)f0.z;
            sW[(cg + 3) * SWS + r] = (__bf16)f0.w;
            sW[(cg + 4) * SWS + r] = (__bf16)f1.x;
            sW[(cg + 5) * SWS + r] = (__bf16)f1.y;
            sW[(cg + 6) * SWS + r] = (__bf16)f1.z;
            sW[(cg + 7) * SWS + r] = (__bf16)f1.w;
        }
#if HAVE_ASYNC_LDS
        __builtin_amdgcn_s_wait_asynccnt(0);
#endif
        __syncthreads();

        if (k0 + BK < K) {
            __builtin_prefetch(W + (size_t)(k0 + BK) * ldw + n0 + (tid & 7) * 8, 0, 1);
            __builtin_prefetch(A + (size_t)(m0 + (tid >> 1)) * lda + k0 + BK, 0, 1);
        }

        // ---- A fragment (16x32 bf16, per ISA layout) ----
        const uint4 a0 = *reinterpret_cast<const uint4*>(&sA[rowA * SAS + kbA]);
        const uint4 a1 = *reinterpret_cast<const uint4*>(&sA[rowA * SAS + kbA + 16]);
        const v16bf af = make_frag(a0, a1);

        // ---- 4 B fragments + WMMA ----
#pragma unroll
        for (int t = 0; t < 4; ++t) {
            const int col = t * 16 + lr;
            const uint4 b0 = *reinterpret_cast<const uint4*>(&sW[col * SWS + kbB]);
            const uint4 b1 = *reinterpret_cast<const uint4*>(&sW[col * SWS + kbB + 16]);
            const v16bf bf = make_frag(b0, b1);
            acc[t] = __builtin_amdgcn_wmma_f32_16x16x32_bf16(
                false, af, false, bf, (short)0, acc[t], false, false);
        }
        __syncthreads();
    }

    // ---- epilogue ----
#pragma unroll
    for (int t = 0; t < 4; ++t) {
        const int col = n0 + t * 16 + lr;
#pragma unroll
        for (int i = 0; i < 8; ++i) {
            const int row = m0 + (wave << 4) + (hi << 3) + i;
            const float v = acc[t][i];
            const size_t off = (size_t)row * ldc + col;
            if (mode == 0) {
                Cf[off] = v;
            } else if (mode == 1) {
                Cf[off] = aux[(size_t)row * ldaux + col] + v;
            } else {
                const float g  = aux[(size_t)row * ldaux + col];
                const float sg = g / (1.f + __expf(-g));
                Cb[off] = (__bf16)(sg * v);
            }
        }
    }
}

// ---------------- RoPE (in-place on qkv fp32 buffer) ----------------
// ref: r=t[:64], im=t[64:]; out[2j]=r*c-im*s; out[2j+1]=r*s+im*c
__global__ __launch_bounds__(256) void rope_kernel(float* __restrict__ qkv) {
    const int m = blockIdx.x;          // b*S + s
    const int s = m % SEQ;
    const int tid = threadIdx.x;
    float* rowp = qkv + (size_t)m * QKVW;
    float rbuf[5], ibuf[5];
    int offs[5], js[5];
#pragma unroll
    for (int i = 0; i < 5; ++i) {
        const int p    = tid + i * 256;     // 0..1279 = (HQ+HKV)*64 pairs
        const int head = p >> 6;
        const int j    = p & 63;
        const int off  = (head < HQ) ? head * HD : DMODEL + (head - HQ) * HD;
        offs[i] = off; js[i] = j;
        rbuf[i] = rowp[off + j];
        ibuf[i] = rowp[off + 64 + j];
    }
    __syncthreads();   // all reads done before interleaved writes
#pragma unroll
    for (int i = 0; i < 5; ++i) {
        const int j = js[i], off = offs[i];
        const float freq = __powf(THETA, -(float)(2 * j) / (float)HD);
        const float ang  = (float)s * freq;
        float sn, cs;
        __sincosf(ang, &sn, &cs);
        rowp[off + 2 * j]     = rbuf[i] * cs - ibuf[i] * sn;
        rowp[off + 2 * j + 1] = rbuf[i] * sn + ibuf[i] * cs;
    }
}

// ---------------- attention: one wave per (b,h,q) row ----------------
__global__ __launch_bounds__(256) void attention_kernel(
    const float* __restrict__ qkv, __bf16* __restrict__ out) {
    const int gw   = blockIdx.x * 8 + (threadIdx.x >> 5);
    const int lane = threadIdx.x & 31;
    const int b    = gw / (HQ * SEQ);
    const int rem  = gw % (HQ * SEQ);
    const int h    = rem / SEQ;
    const int sq   = rem % SEQ;
    const int kh   = h / RATIO;

    const float4 q = *reinterpret_cast<const float4*>(
        qkv + (size_t)(b * SEQ + sq) * QKVW + h * HD + lane * 4);
    const float scale = 0.08838834764831845f;  // 1/sqrt(128)

    float mmax = -3.0e38f, lsum = 0.f;
    float4 accv = {0.f, 0.f, 0.f, 0.f};

    for (int kk = 0; kk <= sq; ++kk) {
        const float* kp = qkv + (size_t)(b * SEQ + kk) * QKVW + DMODEL + kh * HD + lane * 4;
        const float4 kv = *reinterpret_cast<const float4*>(kp);
        float part = q.x * kv.x + q.y * kv.y + q.z * kv.z + q.w * kv.w;
        part += __shfl_xor(part, 16, 32);
        part += __shfl_xor(part, 8, 32);
        part += __shfl_xor(part, 4, 32);
        part += __shfl_xor(part, 2, 32);
        part += __shfl_xor(part, 1, 32);
        const float sc   = part * scale;
        const float nm   = fmaxf(mmax, sc);
        const float corr = __expf(mmax - nm);
        const float p    = __expf(sc - nm);
        mmax = nm;
        lsum = lsum * corr + p;
        const float4 vv = *reinterpret_cast<const float4*>(kp + HKV * HD);
        accv.x = accv.x * corr + p * vv.x;
        accv.y = accv.y * corr + p * vv.y;
        accv.z = accv.z * corr + p * vv.z;
        accv.w = accv.w * corr + p * vv.w;
    }
    const float inv = 1.f / lsum;
    const size_t o = (size_t)(b * SEQ + sq) * DMODEL + h * HD + lane * 4;
    out[o + 0] = (__bf16)(accv.x * inv);
    out[o + 1] = (__bf16)(accv.y * inv);
    out[o + 2] = (__bf16)(accv.z * inv);
    out[o + 3] = (__bf16)(accv.w * inv);
}

// ---------------- host driver ----------------
extern "C" void kernel_launch(void* const* d_in, const int* in_sizes, int n_in,
                              void* d_out, int out_size, void* d_ws, size_t ws_size,
                              hipStream_t stream) {
    const int*   tokens = (const int*)d_in[0];
    const float* emb    = (const float*)d_in[1];
    const float* w_qkv  = (const float*)d_in[2];
    const float* w_o    = (const float*)d_in[3];
    const float* w_gu   = (const float*)d_in[4];
    const float* w_dn   = (const float*)d_in[5];
    const float* ln_a   = (const float*)d_in[6];
    const float* ln_f2  = (const float*)d_in[7];
    const float* ln_fin = (const float*)d_in[8];
    const float* w_lm   = (const float*)d_in[9];
    (void)in_sizes; (void)n_in; (void)out_size; (void)ws_size;

    char* ws = (char*)d_ws;
    float*  xres = (float*)(ws);                          // 16 MB residual fp32
    __bf16* xn   = (__bf16*)(ws + 16777216);              //  8 MB normed bf16
    float*  qkv  = (float*)(ws + 25165824);               // 24 MB qkv fp32
    __bf16* hbuf = (__bf16*)(ws + 50331648);              // 32 MB attn-out / ffn-h bf16
    float*  gate = (float*)(ws + 83886080);               // 64 MB gate fp32

    const dim3 blk(256);
    const dim3 gQKV(QKVW / BN, MTOT / BM);
    const dim3 gO(DMODEL / BN, MTOT / BM);
    const dim3 gP(PFFN / BN, MTOT / BM);
    const dim3 gV(VOCAB / BN, MTOT / BM);

    embed_kernel<<<MTOT, blk, 0, stream>>>(tokens, emb, xres);

    for (int l = 0; l < LAYERS; ++l) {
        // ---- attention block ----
        rmsnorm_kernel<<<MTOT, blk, 0, stream>>>(xres, ln_a + (size_t)l * DMODEL, xn);
        gemm_bf16_wmma<<<gQKV, blk, 0, stream>>>(
            xn, DMODEL, w_qkv + (size_t)l * DMODEL * QKVW, QKVW,
            qkv, nullptr, QKVW, DMODEL, 0, nullptr, 0);
        rope_kernel<<<MTOT, blk, 0, stream>>>(qkv);
        attention_kernel<<<(BATCH * SEQ * HQ) / 8, blk, 0, stream>>>(qkv, hbuf);
        gemm_bf16_wmma<<<gO, blk, 0, stream>>>(
            hbuf, DMODEL, w_o + (size_t)l * DMODEL * DMODEL, DMODEL,
            xres, nullptr, DMODEL, DMODEL, 1, xres, DMODEL);

        // ---- ffn block ----
        rmsnorm_kernel<<<MTOT, blk, 0, stream>>>(xres, ln_f2 + (size_t)l * DMODEL, xn);
        gemm_bf16_wmma<<<gP, blk, 0, stream>>>(
            xn, DMODEL, w_gu + (size_t)l * DMODEL * (2 * PFFN), 2 * PFFN,
            gate, nullptr, PFFN, DMODEL, 0, nullptr, 0);
        gemm_bf16_wmma<<<gP, blk, 0, stream>>>(
            xn, DMODEL, w_gu + (size_t)l * DMODEL * (2 * PFFN) + PFFN, 2 * PFFN,
            nullptr, hbuf, PFFN, DMODEL, 2, gate, PFFN);
        gemm_bf16_wmma<<<gO, blk, 0, stream>>>(
            hbuf, PFFN, w_dn + (size_t)l * PFFN * DMODEL, DMODEL,
            xres, nullptr, DMODEL, PFFN, 1, xres, DMODEL);
    }

    // ---- final norm + LM head ----
    rmsnorm_kernel<<<MTOT, blk, 0, stream>>>(xres, ln_fin, xn);
    gemm_bf16_wmma<<<gV, blk, 0, stream>>>(
        xn, DMODEL, w_lm, VOCAB,
        (float*)d_out, nullptr, VOCAB, DMODEL, 0, nullptr, 0);
}
